// DynamicNestedHierarchy_64098091926042
// MI455X (gfx1250) — matrix operand order repaired and checked by reading, not compiled, not run on hardware
//
#include <hip/hip_runtime.h>
#include <hip/hip_bf16.h>

// ---------------------------------------------------------------------------
// Types for CDNA5 WMMA
// ---------------------------------------------------------------------------
typedef __attribute__((ext_vector_type(16))) __bf16 v16bf;
typedef __attribute__((ext_vector_type(8)))  float  v8f;
typedef __attribute__((ext_vector_type(4)))  int    v4i;

union U16 { v16bf v; v4i q[2]; };

__device__ __forceinline__ v8f wmma_bf16(v16bf a, v16bf b, v8f c) {
  // D = A(16x32 bf16) * B(32x16 bf16) + C(16x16 f32)
  return __builtin_amdgcn_wmma_f32_16x16x32_bf16(
      /*neg_a=*/false, a, /*neg_b=*/false, b,
      /*c_mod=*/(short)0, c, /*reuse_a=*/false, /*reuse_b=*/false);
}

// ---------------------------------------------------------------------------
// gfx1250 async global->LDS copy (ASYNCcnt-tracked), 16 bytes per lane.
// vdst VGPR carries the per-lane LDS byte offset; generic pointers to
// __shared__ carry the LDS offset in their low 32 bits (aperture mapping).
// ---------------------------------------------------------------------------
__device__ __forceinline__ unsigned lds_off(const void* p) {
  return (unsigned)(uintptr_t)p;
}
__device__ __forceinline__ void async_copy_b128(unsigned ldsOff, const void* g) {
  unsigned long long ga = (unsigned long long)g;
  asm volatile("global_load_async_to_lds_b128 %0, %1, off"
               :: "v"(ldsOff), "v"(ga)
               : "memory");
}
__device__ __forceinline__ void wait_async0() {
  asm volatile("s_wait_asynccnt 0" ::: "memory");
}

// ---------------------------------------------------------------------------
// Math helpers (jax.nn.gelu default = tanh approximation)
// ---------------------------------------------------------------------------
__device__ __forceinline__ float geluf(float x) {
  const float c0 = 0.7978845608028654f, c1 = 0.044715f;
  float x3 = x * x * x;
  float t  = tanhf(c0 * (x + c1 * x3));
  return 0.5f * x * (1.f + t);
}
__device__ __forceinline__ float gelu_grad(float x) {
  const float c0 = 0.7978845608028654f, c1 = 0.044715f;
  float x2 = x * x;
  float u  = c0 * (x + c1 * x2 * x);
  float t  = tanhf(u);
  float du = c0 * (1.f + 3.f * c1 * x2);
  return 0.5f * (1.f + t) + 0.5f * x * (1.f - t * t) * du;
}

// ---------------------------------------------------------------------------
// Generic bf16 WMMA GEMM:  C[M,N] = A[M,K] (row-major) * B'[K,N]
// where B is stored as [N,K] row-major (i.e. B'[k][n] = B[n*K + k]).
// Block tile 128x128x32, 256 threads = 8 waves (2 along M x 4 along N),
// each wave computes 64x32 via 4x2 WMMA 16x16 fragments.
// LDS tiles are double-buffered; staging uses global_load_async_to_lds_b128
// so the async DMA for tile t+1 overlaps the WMMA burst for tile t.
// Split-K via gridDim.z (kLen = K / splits); epilogue 7 accumulates
// atomically into an f32 buffer.
// ---------------------------------------------------------------------------
#define GEMM_PITCH 40  // halfs per LDS row (32 data + 8 pad), 80B, 16B aligned

__global__ __launch_bounds__(256) void gemm_bf16_wmma(
    const __bf16* __restrict__ A,   // [M,K]
    const __bf16* __restrict__ B,   // [Nc,K]
    int M, int Nc, int K, int kLen, int epi,
    __bf16* __restrict__ Crow,      // bf16 row-major out
    __bf16* __restrict__ Ct,        // bf16 transposed out [Nc,M]
    float*  __restrict__ Cf,        // f32 out / atomic accumulation target
    const float*  __restrict__ bias,
    const float*  __restrict__ Vf,  // ERR: target values (f32 row-major)
    const __bf16* __restrict__ Pre, // DPRE: pre-activations (bf16 row-major)
    __bf16* __restrict__ Pout,      // PRE_HK: pre-activation store
    float* __restrict__ lossAcc,    // ERR: sum(err^2) accumulator
    float invBN) {
  __shared__ __bf16 As[2][128 * GEMM_PITCH];
  __shared__ __bf16 Bs[2][128 * GEMM_PITCH];

  const int tid  = threadIdx.x;
  const size_t bM = (size_t)blockIdx.y * 128;
  const size_t bN = (size_t)blockIdx.x * 128;
  const int kBeg = blockIdx.z * kLen;
  const int nT   = kLen >> 5;  // K-tiles of 32

  // Staging: each thread moves 2x16B of A and 2x16B of B per tile.
  const int sRow = tid >> 2;           // 0..63
  const int sCol = (tid & 3) << 3;     // 0,8,16,24 (halfs)
  const __bf16* Aptr = A + (bM + sRow) * (size_t)K + kBeg + sCol;
  const __bf16* Bptr = B + (bN + sRow) * (size_t)K + kBeg + sCol;

  // Per-thread LDS byte offsets for the 4 chunks (buffer 0); buffer 1 is
  // +bufStride bytes inside each [2][...] array.
  const unsigned la0 = lds_off(&As[0][sRow * GEMM_PITCH + sCol]);
  const unsigned la1 = lds_off(&As[0][(sRow + 64) * GEMM_PITCH + sCol]);
  const unsigned lb0 = lds_off(&Bs[0][sRow * GEMM_PITCH + sCol]);
  const unsigned lb1 = lds_off(&Bs[0][(sRow + 64) * GEMM_PITCH + sCol]);
  const unsigned bufStride = (unsigned)(128 * GEMM_PITCH * sizeof(__bf16));

  auto issueAsync = [&](int t, int buf) {
    const size_t off = (size_t)t * 32;
    const unsigned bo_ = (unsigned)buf * bufStride;
    async_copy_b128(la0 + bo_, Aptr + off);
    async_copy_b128(la1 + bo_, Aptr + off + (size_t)64 * K);
    async_copy_b128(lb0 + bo_, Bptr + off);
    async_copy_b128(lb1 + bo_, Bptr + off + (size_t)64 * K);
  };

  const int lane = tid & 31;
  const int wv   = tid >> 5;
  const int wM   = (wv >> 2) << 6;   // 0 / 64
  const int wN   = (wv & 3) << 5;    // 0 / 32 / 64 / 96
  const int l16  = lane & 15;
  const int lh   = lane >> 4;        // 0 / 1

  v8f acc[4][2];
#pragma unroll
  for (int m = 0; m < 4; ++m)
#pragma unroll
    for (int n = 0; n < 2; ++n)
#pragma unroll
      for (int r = 0; r < 8; ++r) acc[m][n][r] = 0.f;

  issueAsync(0, 0);
  wait_async0();
  __syncthreads();

  for (int t = 0; t < nT; ++t) {
    const int cur = t & 1;
    if (t + 1 < nT) issueAsync(t + 1, (t + 1) & 1);  // DMA under the WMMAs

    const __bf16* as = As[cur];
    const __bf16* bs = Bs[cur];

    // A fragments: 16-bit A 16x32 ISA layout:
    // lanes 0-15 (M=l16): K 0..7 in V0-3, K 16..23 in V4-7
    // lanes 16-31        : K 8..15      , K 24..31
    v16bf af[4];
#pragma unroll
    for (int m = 0; m < 4; ++m) {
      const __bf16* p = as + (wM + m * 16 + l16) * GEMM_PITCH + (lh << 3);
      U16 u;
      u.q[0] = *(const v4i*)p;
      u.q[1] = *(const v4i*)(p + 16);
      af[m] = u.v;
    }
    // B fragments: 32x16, same K order as A: lanes 0-15 (N=l16) K 0..7, 16..23;
    // lanes 16-31 K 8..15, 24..31. LDS B tile is N-major so K is contiguous.
    v16bf bfr[2];
#pragma unroll
    for (int n = 0; n < 2; ++n) {
      const __bf16* p = bs + (wN + n * 16 + l16) * GEMM_PITCH + (lh << 3);
      U16 u;
      u.q[0] = *(const v4i*)p;
      u.q[1] = *(const v4i*)(p + 16);
      bfr[n] = u.v;
    }
#pragma unroll
    for (int n = 0; n < 2; ++n)
#pragma unroll
      for (int m = 0; m < 4; ++m)
        acc[m][n] = wmma_bf16(af[m], bfr[n], acc[m][n]);

    if (t + 1 < nT) wait_async0();  // own async copies into next buffer done
    __syncthreads();
  }

  // ---------------- Epilogue ----------------
  // C f32 16x16 layout: VGPR r, lanes 0-15 -> M=r, lanes 16-31 -> M=r+8; N=l16.
  float lsum = 0.f;
#pragma unroll
  for (int m = 0; m < 4; ++m)
#pragma unroll
    for (int n = 0; n < 2; ++n) {
      const size_t row0 = bM + wM + m * 16 + lh * 8;
      const size_t col  = bN + wN + n * 16 + l16;
#pragma unroll
      for (int r = 0; r < 8; ++r) {
        const float  val = acc[m][n][r];
        const size_t row = row0 + r;
        const size_t idx = row * (size_t)Nc + col;
        switch (epi) {
          case 0:  // bf16 row-major
            Crow[idx] = (__bf16)val;
            break;
          case 1:  // bf16 row-major + transposed copy
            Crow[idx] = (__bf16)val;
            Ct[col * (size_t)M + row] = (__bf16)val;
            break;
          case 2:  // f32 (+ optional bias)
            Cf[idx] = val + (bias ? bias[col] : 0.f);
            break;
          case 3:  // gelu, bf16 row-major
            Crow[idx] = (__bf16)geluf(val);
            break;
          case 4: {  // pre + gelu(pre) row-major + gelu(pre) transposed
            Pout[idx] = (__bf16)val;
            const float g = geluf(val);
            Crow[idx] = (__bf16)g;
            Ct[col * (size_t)M + row] = (__bf16)g;
            break;
          }
          case 5: {  // err = (m - v); store errn = err/BN both layouts; sum e^2
            const float e  = val - Vf[idx];
            lsum += e * e;
            const float en = e * invBN;
            Crow[idx] = (__bf16)en;
            Ct[col * (size_t)M + row] = (__bf16)en;
            break;
          }
          case 6: {  // dpre = dhk * gelu'(pre); store transposed only
            const float p = (float)Pre[idx];
            Ct[col * (size_t)M + row] = (__bf16)(val * gelu_grad(p));
            break;
          }
          case 7:  // split-K accumulation
            atomicAdd(&Cf[idx], val);
            break;
          default: break;
        }
      }
    }
  if (epi == 5) atomicAdd(lossAcc, lsum);
}

// ---------------------------------------------------------------------------
// Small helper kernels
// ---------------------------------------------------------------------------
__global__ void cvt_f32_bf16(const float* __restrict__ s, __bf16* __restrict__ d, int n) {
  int j = blockIdx.x * blockDim.x + threadIdx.x;
  if (j < n) d[j] = (__bf16)s[j];
}
// dst[C,R] bf16 = transpose of src[R,C] f32
__global__ void cvtT_f32_bf16(const float* __restrict__ s, __bf16* __restrict__ d, int R, int C) {
  int j = blockIdx.x * blockDim.x + threadIdx.x;
  if (j < R * C) {
    int r = j / C, c = j % C;
    d[(size_t)c * R + r] = (__bf16)s[j];
  }
}
__global__ void zero_f32(float* __restrict__ p, int n) {
  int j = blockIdx.x * blockDim.x + threadIdx.x;
  if (j < n) p[j] = 0.f;
}
// scal[0..L-1]=eta, scal[L..2L-1]=alpha, scal[2L..3L-1]=0 (loss accumulators)
__global__ void scalars_kernel(const float* __restrict__ eta_base,
                               const float* __restrict__ alpha_base,
                               float* __restrict__ scal, int D, int L) {
  __shared__ float red[256];
  const int tid = threadIdx.x;
  for (int l = 0; l < L; ++l) {
    float s = 0.f;
    for (int j = tid; j < D; j += 256) {
      float x = eta_base[l * D + j];
      s += fmaxf(x, 0.f) + log1pf(expf(-fabsf(x)));  // stable softplus
    }
    red[tid] = s; __syncthreads();
    for (int o = 128; o > 0; o >>= 1) { if (tid < o) red[tid] += red[tid + o]; __syncthreads(); }
    if (tid == 0) scal[l] = red[0] / D;
    __syncthreads();
    s = 0.f;
    for (int j = tid; j < D; j += 256) {
      float x = alpha_base[l * D + j];
      s += 1.f / (1.f + expf(-x));
    }
    red[tid] = s; __syncthreads();
    for (int o = 128; o > 0; o >>= 1) { if (tid < o) red[tid] += red[tid + o]; __syncthreads(); }
    if (tid == 0) scal[L + l] = red[0] / D;
    __syncthreads();
  }
  if (tid < L) scal[2 * L + tid] = 0.f;
}
// new_w = gate ? (1-alpha)*w - eta*clip(g,±1) : w
__global__ void wupdate_kernel(const float* __restrict__ g, const float* __restrict__ w,
                               float* __restrict__ out, int n,
                               const float* __restrict__ scal, int lvl, int L) {
  const float eta   = scal[lvl];
  const float alpha = scal[L + lvl];
  const bool  gate  = scal[2 * L + lvl] > 0.f;
  int j = blockIdx.x * blockDim.x + threadIdx.x;
  if (j < n) {
    float gv = fminf(fmaxf(g[j], -1.f), 1.f);
    float wv = w[j];
    out[j] = gate ? (1.f - alpha) * wv - eta * gv : wv;
  }
}

// ---------------------------------------------------------------------------
// Host launcher
// ---------------------------------------------------------------------------
static inline void launch_gemm(hipStream_t s, const __bf16* A, const __bf16* B,
                               int M, int Nc, int K, int splits, int epi,
                               __bf16* Crow, __bf16* Ct, float* Cf,
                               const float* bias, const float* Vf,
                               const __bf16* Pre, __bf16* Pout,
                               float* lossAcc, float invBN) {
  dim3 grid(Nc / 128, M / 128, splits);
  gemm_bf16_wmma<<<grid, 256, 0, s>>>(A, B, M, Nc, K, K / splits, epi,
                                      Crow, Ct, Cf, bias, Vf, Pre, Pout,
                                      lossAcc, invBN);
}
static inline int nblk(int n) { return (n + 255) / 256; }

extern "C" void kernel_launch(void* const* d_in, const int* in_sizes, int n_in,
                              void* d_out, int out_size, void* d_ws, size_t ws_size,
                              hipStream_t stream) {
  (void)in_sizes; (void)n_in; (void)out_size; (void)ws_size;
  const int   D = 384, H = 768, L = 2, SPLITS = 32;
  const size_t BN = 65536;  // 8 * 8192
  const float invBN = 1.f / 65536.f;

  const float* x     = (const float*)d_in[0];
  const float* Wq    = (const float*)d_in[1];
  const float* Wk    = (const float*)d_in[2];
  const float* Wv    = (const float*)d_in[3];
  const float* Wo    = (const float*)d_in[4];
  const float* bo    = (const float*)d_in[5];
  const float* w1s   = (const float*)d_in[6];
  const float* w2s   = (const float*)d_in[7];
  const float* etab  = (const float*)d_in[8];
  const float* alphb = (const float*)d_in[9];

  float* outF   = (float*)d_out;                 // [BN, D]
  float* out_w1 = outF + BN * D;                 // [L, H, D]
  float* out_w2 = out_w1 + (size_t)L * H * D;    // [L, D, H]

  // ---- workspace partition ----
  char* wp = (char*)d_ws;
  auto bump = [&](size_t bytes) -> char* {
    char* p = wp;
    wp += (bytes + 255) & ~(size_t)255;
    return p;
  };
  float*  scal  = (float*)bump(64);
  __bf16* Wqb   = (__bf16*)bump((size_t)D * D * 2);
  __bf16* Wkb   = (__bf16*)bump((size_t)D * D * 2);
  __bf16* Wvb   = (__bf16*)bump((size_t)D * D * 2);
  __bf16* Wob   = (__bf16*)bump((size_t)D * D * 2);
  __bf16* w1b   = (__bf16*)bump((size_t)L * H * D * 2);
  __bf16* w2b   = (__bf16*)bump((size_t)L * D * H * 2);
  __bf16* w2Tb  = (__bf16*)bump((size_t)L * H * D * 2);
  float*  gbuf  = (float*)bump((size_t)H * D * 4);
  __bf16* Xbf   = (__bf16*)bump(BN * D * 2);
  __bf16* Qbf   = (__bf16*)bump(BN * D * 2);
  __bf16* Y1    = (__bf16*)bump(BN * D * 2);
  __bf16* Y2    = (__bf16*)bump(BN * D * 2);
  __bf16* Kbf   = (__bf16*)bump(BN * D * 2);
  __bf16* KT    = (__bf16*)bump(BN * D * 2);
  __bf16* Errn  = (__bf16*)bump(BN * D * 2);
  __bf16* ErrnT = (__bf16*)bump(BN * D * 2);
  float*  Vf32  = (float*)bump(BN * D * 4);
  __bf16* Hy    = (__bf16*)bump(BN * H * 2);
  __bf16* PreB  = (__bf16*)bump(BN * H * 2);
  __bf16* Hk    = (__bf16*)bump(BN * H * 2);
  __bf16* HkT   = (__bf16*)bump(BN * H * 2);
  __bf16* DpreT = (__bf16*)bump(BN * H * 2);

  // ---- scalars + conversions ----
  scalars_kernel<<<1, 256, 0, stream>>>(etab, alphb, scal, D, L);
  cvt_f32_bf16<<<nblk(BN * D), 256, 0, stream>>>(x,  Xbf, (int)(BN * D));
  cvt_f32_bf16<<<nblk(D * D), 256, 0, stream>>>(Wq, Wqb, D * D);
  cvt_f32_bf16<<<nblk(D * D), 256, 0, stream>>>(Wk, Wkb, D * D);
  cvt_f32_bf16<<<nblk(D * D), 256, 0, stream>>>(Wv, Wvb, D * D);
  cvt_f32_bf16<<<nblk(D * D), 256, 0, stream>>>(Wo, Wob, D * D);
  cvt_f32_bf16<<<nblk(L * H * D), 256, 0, stream>>>(w1s, w1b, L * H * D);
  cvt_f32_bf16<<<nblk(L * D * H), 256, 0, stream>>>(w2s, w2b, L * D * H);
  for (int i = 0; i < L; ++i)
    cvtT_f32_bf16<<<nblk(D * H), 256, 0, stream>>>(w2s + (size_t)i * D * H,
                                                   w2Tb + (size_t)i * H * D, D, H);

  // ---- projections ----
  // q (initial y), bf16
  launch_gemm(stream, Xbf, Wqb, (int)BN, D, D, 1, 0, Qbf, nullptr, nullptr,
              nullptr, nullptr, nullptr, nullptr, nullptr, invBN);
  // k, bf16 row + transposed
  launch_gemm(stream, Xbf, Wkb, (int)BN, D, D, 1, 1, Kbf, KT, nullptr,
              nullptr, nullptr, nullptr, nullptr, nullptr, invBN);
  // v, f32
  launch_gemm(stream, Xbf, Wvb, (int)BN, D, D, 1, 2, nullptr, nullptr, Vf32,
              nullptr, nullptr, nullptr, nullptr, nullptr, invBN);

  // ---- levels ----
  for (int i = 0; i < L; ++i) {
    const __bf16* yin  = (i == 0) ? Qbf : Y1;
    __bf16*       yout = (i == 0) ? Y1 : Y2;
    const __bf16* w1p  = w1b + (size_t)i * H * D;   // [H,D] = [N,K]
    const __bf16* w2p  = w2b + (size_t)i * D * H;   // [D,H] = [N,K]
    const __bf16* w2Tp = w2Tb + (size_t)i * H * D;  // [H,D] = [N,K]
    float* lossp = scal + 2 * L + i;

    // read path: y = gelu(y @ w1^T) @ w2^T
    launch_gemm(stream, yin, w1p, (int)BN, H, D, 1, 3, Hy, nullptr, nullptr,
                nullptr, nullptr, nullptr, nullptr, nullptr, invBN);
    launch_gemm(stream, Hy, w2p, (int)BN, D, H, 1, 0, yout, nullptr, nullptr,
                nullptr, nullptr, nullptr, nullptr, nullptr, invBN);

    // loss path: pre = k @ w1^T; hk = gelu(pre) (row + transposed)
    launch_gemm(stream, Kbf, w1p, (int)BN, H, D, 1, 4, Hk, HkT, nullptr,
                nullptr, nullptr, nullptr, PreB, nullptr, invBN);
    // m = hk @ w2^T; errn = (m - v)/BN (row + transposed); accumulate sum(e^2)
    launch_gemm(stream, Hk, w2p, (int)BN, D, H, 1, 5, Errn, ErrnT, nullptr,
                nullptr, Vf32, nullptr, nullptr, lossp, invBN);
    // dpreT = (errn @ w2) * gelu'(pre), stored [H,BN]
    launch_gemm(stream, Errn, w2Tp, (int)BN, H, D, 1, 6, nullptr, DpreT,
                nullptr, nullptr, nullptr, PreB, nullptr, nullptr, invBN);

    // g2 = errn^T @ hk : [D,H], split-K over BN
    zero_f32<<<nblk(D * H), 256, 0, stream>>>(gbuf, D * H);
    launch_gemm(stream, ErrnT, HkT, D, H, (int)BN, SPLITS, 7, nullptr, nullptr,
                gbuf, nullptr, nullptr, nullptr, nullptr, nullptr, invBN);
    wupdate_kernel<<<nblk(D * H), 256, 0, stream>>>(
        gbuf, w2s + (size_t)i * D * H, out_w2 + (size_t)i * D * H, D * H, scal, i, L);

    // g1 = dpre^T @ k : [H,D], split-K over BN
    zero_f32<<<nblk(H * D), 256, 0, stream>>>(gbuf, H * D);
    launch_gemm(stream, DpreT, KT, H, D, (int)BN, SPLITS, 7, nullptr, nullptr,
                gbuf, nullptr, nullptr, nullptr, nullptr, nullptr, invBN);
    wupdate_kernel<<<nblk(H * D), 256, 0, stream>>>(
        gbuf, w1s + (size_t)i * H * D, out_w1 + (size_t)i * H * D, H * D, scal, i, L);
  }

  // ---- output projection (+ bias), f32 ----
  launch_gemm(stream, Y2, Wob, (int)BN, D, D, 1, 2, nullptr, nullptr, outF,
              bo, nullptr, nullptr, nullptr, nullptr, invBN);
}
